// TPGStep_71347996721425
// MI455X (gfx1250) — matrix operand order repaired and checked by reading, not compiled, not run on hardware
//
#include <hip/hip_runtime.h>

// ---------------------------------------------------------------------------
// Problem constants (from reference)
// ---------------------------------------------------------------------------
#define BB 4
#define TT 2048
#define VV 1024
#define SS 512
#define II 4096
#define BT (BB * TT)
#define EPSR 1.1920929e-07f

typedef __attribute__((ext_vector_type(16))) __bf16 v16bf;
typedef __attribute__((ext_vector_type(8)))  float  v8f;
typedef __attribute__((ext_vector_type(4)))  unsigned int u32x4;
typedef __attribute__((ext_vector_type(8)))  unsigned int u32x8;

// ---------------------------------------------------------------------------
// Tensor Data Mover: 2-D tile load  global -> LDS  (CDNA5 TDM, ISA ch.7/8 of
// async doc).  data_size = 2B (bf16).  pad_enable with pad_interval=32 DWORDs
// and pad_amount=8 DWORDs  =>  each 64-elem (128B) row is followed by 32B of
// padding: LDS row stride = 160B = 80 bf16 elems (16B aligned, bank-staggered).
// Issued once per wave (TDM ignores EXEC); tracked by TENSORcnt.
// ---------------------------------------------------------------------------
#define KT   64   // K tile (elements)
#define LDSR 80   // padded LDS row stride in bf16 elems (matches TDM pad)

__device__ __forceinline__ void tdm_load_tile(unsigned lds_addr,
                                              const __bf16* gaddr,
                                              unsigned tensor_d0,   // K (elems)
                                              unsigned tensor_d1,   // rows
                                              unsigned long long stride_elems)
{
    union { unsigned long long q[2]; u32x4 v; } g0;
    const unsigned long long ga = (unsigned long long)gaddr;
    // group0: count=1 | lds_addr<<32 ; global_addr[56:0] | type=2 <<126
    g0.q[0] = 1ull | ((unsigned long long)lds_addr << 32);
    g0.q[1] = (ga & 0x01FFFFFFFFFFFFFFull) | (2ull << 62);

    union { unsigned long long q[4]; u32x8 v; } g1;
    const unsigned long long flags =
        (1ull << 16)     // data_size = 2 bytes
      | (1ull << 20)     // pad_enable
      | (4ull << 22)     // pad_interval: 32 DWORDs
      | (7ull << 25);    // pad_amount:   8 DWORDs
    g1.q[0] = flags | ((unsigned long long)(tensor_d0 & 0xFFFFu) << 48);
    g1.q[1] = ((unsigned long long)tensor_d0 >> 16)
            | ((unsigned long long)tensor_d1 << 16)
            | ((unsigned long long)KT << 48);              // tile_dim0 = 64
    g1.q[2] = 128ull                                        // tile_dim1 = 128 rows
            | ((stride_elems & 0xFFFFFFFFull) << 32);       // dim0_stride lo
    g1.q[3] = (stride_elems >> 32) & 0xFFFFull;             // dim0_stride hi

    asm volatile("tensor_load_to_lds %0, %1" :: "s"(g0.v), "s"(g1.v) : "memory");
}

__device__ __forceinline__ unsigned lds_addr_of(const void* p)
{
    return (unsigned)(unsigned long long)p;   // low 32 bits = LDS byte address
}

// ---------------------------------------------------------------------------
// Generic WMMA bf16 GEMM:  C[M,N] = A[M,K] * op(B)
//   BKN == false : B stored [N,K] row-major (C = A * B^T)  -> A,B via TDM
//   BKN == true  : B stored [K,N] row-major (C = A * B)    -> A via TDM,
//                  B transposed into LDS manually.
// Batched over z = n*ZB + b with independent (n,b) strides for A/B/C.
// EPI: 0 = fp32 store, 1 = bf16 store, 2 = decay-weighted bf16 store (scores)
// Block: 256 threads (8 waves). C tile 128x128; wave tile 32x64 (2x4 WMMA).
// K tile 64 (two 16x16x32 sub-steps), double-buffered LDS fed by TDM.
// M, N multiples of 128; K multiple of 64 (true for every call here).
// ---------------------------------------------------------------------------
template <int EPI, bool BKN>
__global__ __launch_bounds__(256)
void gemm_wmma(const __bf16* __restrict__ A, const __bf16* __restrict__ B,
               void* __restrict__ C, int M, int N, int K,
               long sAn, long sAb, long sBn, long sBb, long sCn, long sCb,
               int ZB, const float* __restrict__ decay_logits)
{
    __shared__ __bf16 As[2][128 * LDSR];
    __shared__ __bf16 Bs[2][128 * LDSR];

    const int z  = blockIdx.z;
    const int nz = z / ZB;
    const int bz = z % ZB;
    A += nz * sAn + bz * sAb;
    B += nz * sBn + bz * sBb;
    const long coff = nz * sCn + bz * sCb;

    const int m0 = blockIdx.y * 128;
    const int n0 = blockIdx.x * 128;

    const int lane = threadIdx.x & 31;
    const int wave = threadIdx.x >> 5;
    const int wr   = (wave >> 1) * 32;   // wave row offset inside 128x128 tile
    const int wc   = (wave & 1) * 64;    // wave col offset
    const int lm   = lane & 15;
    const int half = lane >> 4;
    const bool w0  = (wave == 0);

    v8f acc[2][4];
#pragma unroll
    for (int mt = 0; mt < 2; ++mt)
#pragma unroll
        for (int nt = 0; nt < 4; ++nt)
#pragma unroll
            for (int r = 0; r < 8; ++r) acc[mt][nt][r] = 0.0f;

    const int nk = K / KT;

    // prologue: DMA tile 0
    if (w0) {
        tdm_load_tile(lds_addr_of(&As[0][0]), A + (long)m0 * K,
                      (unsigned)K, (unsigned)M, (unsigned long long)K);
        if constexpr (!BKN)
            tdm_load_tile(lds_addr_of(&Bs[0][0]), B + (long)n0 * K,
                          (unsigned)K, (unsigned)N, (unsigned long long)K);
    }

    for (int kt = 0; kt < nk; ++kt) {
        const int cur = kt & 1;
        const int k0  = kt * KT;

        __syncthreads();   // all waves finished reading buffers of tile kt-1

        if constexpr (BKN) {
            // stage B tile [K=64 x N=128] transposed -> Bs[n][k], coalesced reads
#pragma unroll 4
            for (int i = 0; i < 32; ++i) {
                const int e  = threadIdx.x + i * 256;   // 0..8191
                const int cc = e & 127;                 // n within tile
                const int kk = e >> 7;                  // k within tile
                Bs[cur][cc * LDSR + kk] = B[(long)(k0 + kk) * N + n0 + cc];
            }
        }
        if (w0) __builtin_amdgcn_s_wait_tensorcnt(0);   // tile kt DMA complete

        __syncthreads();   // tile kt data visible to all waves

        // overlap: kick off DMA for tile kt+1 into the other buffer
        if (w0 && (kt + 1 < nk)) {
            tdm_load_tile(lds_addr_of(&As[cur ^ 1][0]),
                          A + (long)m0 * K + (k0 + KT),
                          (unsigned)K, (unsigned)M, (unsigned long long)K);
            if constexpr (!BKN)
                tdm_load_tile(lds_addr_of(&Bs[cur ^ 1][0]),
                              B + (long)n0 * K + (k0 + KT),
                              (unsigned)K, (unsigned)N, (unsigned long long)K);
        }

        const __bf16* as = &As[cur][0];
        const __bf16* bs = &Bs[cur][0];
#pragma unroll
        for (int ks = 0; ks < 2; ++ks) {
            // ---- fragments (ISA 7.12.2 bf16 layouts) ----
            v16bf af[2], bf[4];
#pragma unroll
            for (int mt = 0; mt < 2; ++mt) {
                const int row = wr + mt * 16 + lm;
#pragma unroll
                for (int v = 0; v < 8; ++v) {
                    const int kk = ((v & 4) ? 16 : 0) + half * 8 + (v & 3) * 2;
                    af[mt][2 * v]     = as[row * LDSR + ks * 32 + kk];
                    af[mt][2 * v + 1] = as[row * LDSR + ks * 32 + kk + 1];
                }
            }
#pragma unroll
            for (int nt = 0; nt < 4; ++nt) {
                const int col = wc + nt * 16 + lm;
#pragma unroll
                for (int v = 0; v < 8; ++v) {
                    const int kk = ((v & 4) ? 16 : 0) + half * 8 + (v & 3) * 2;
                    bf[nt][2 * v]     = bs[col * LDSR + ks * 32 + kk];
                    bf[nt][2 * v + 1] = bs[col * LDSR + ks * 32 + kk + 1];
                }
            }
#pragma unroll
            for (int mt = 0; mt < 2; ++mt)
#pragma unroll
                for (int nt = 0; nt < 4; ++nt)
                    acc[mt][nt] = __builtin_amdgcn_wmma_f32_16x16x32_bf16(
                        false, af[mt], false, bf[nt],
                        (short)0, acc[mt][nt], false, false);
        }
    }

    // ---- epilogue ----
    float decay = 0.0f;
    if (EPI == 2) decay = 1.0f / (1.0f + expf(-decay_logits[nz]));

    float* Cf  = (float*)C;
    __bf16* Cb = (__bf16*)C;
#pragma unroll
    for (int mt = 0; mt < 2; ++mt) {
#pragma unroll
        for (int nt = 0; nt < 4; ++nt) {
            const int colg = n0 + wc + nt * 16 + lm;
#pragma unroll
            for (int r = 0; r < 8; ++r) {
                const int rowg = m0 + wr + mt * 16 + half * 8 + r;
                const long idx = coff + (long)rowg * N + colg;
                const float cv = acc[mt][nt][r];
                if (EPI == 0) {
                    Cf[idx] = cv;
                } else if (EPI == 1) {
                    Cb[idx] = (__bf16)cv;
                } else {
                    // decay mask: w = (s>t) ? decay^max(s-t-1,0) : 0
                    const int d = colg - rowg;
                    float w = 0.0f;
                    if (d > 0) w = powf(decay, (float)(d - 1));
                    Cb[idx] = (__bf16)(cv * w);
                }
            }
        }
    }
}

// ---------------------------------------------------------------------------
// Elementwise / reduction kernels
// ---------------------------------------------------------------------------
__global__ __launch_bounds__(256)
void rmsnorm_bf16_kernel(const float* __restrict__ x, __bf16* __restrict__ xn)
{
    __shared__ float red[256];
    const long row = blockIdx.x;
    const float* xr = x + row * VV;
    float s = 0.0f;
    for (int i = threadIdx.x; i < VV; i += 256) { const float v = xr[i]; s += v * v; }
    red[threadIdx.x] = s;
    __syncthreads();
    for (int o = 128; o > 0; o >>= 1) {
        if ((int)threadIdx.x < o) red[threadIdx.x] += red[threadIdx.x + o];
        __syncthreads();
    }
    const float r = rsqrtf(red[0] / (float)VV + EPSR);
    for (int i = threadIdx.x; i < VV; i += 256) xn[row * VV + i] = (__bf16)(xr[i] * r);
}

__global__ void f32_to_bf16_kernel(const float* __restrict__ in, __bf16* __restrict__ out, long n)
{
    for (long i = blockIdx.x * (long)blockDim.x + threadIdx.x; i < n;
         i += (long)gridDim.x * blockDim.x)
        out[i] = (__bf16)in[i];
}

__global__ __launch_bounds__(128)
void scale_gate_kernel(const __bf16* __restrict__ xn, const float* __restrict__ w,
                       float* __restrict__ out)
{
    __shared__ float red[3][128];
    const int tid = threadIdx.x;
    const long row = blockIdx.x;
    const __bf16* xr = xn + row * VV;
    float s[3] = {0.f, 0.f, 0.f};
    for (int i = tid; i < VV; i += 128) {
        const float xv = (float)xr[i];
        s[0] += xv * w[i];
        s[1] += xv * w[VV + i];
        s[2] += xv * w[2 * VV + i];
    }
    for (int k = 0; k < 3; ++k) red[k][tid] = s[k];
    __syncthreads();
    for (int o = 64; o > 0; o >>= 1) {
        if (tid < o) for (int k = 0; k < 3; ++k) red[k][tid] += red[k][tid + o];
        __syncthreads();
    }
    if (tid == 0) {
        const float m = fmaxf(red[0][0], fmaxf(red[1][0], red[2][0]));
        const float e0 = expf(red[0][0] - m), e1 = expf(red[1][0] - m), e2 = expf(red[2][0] - m);
        const float inv = 1.0f / (e0 + e1 + e2);
        out[row * 3 + 0] = e0 * inv;
        out[row * 3 + 1] = e1 * inv;
        out[row * 3 + 2] = e2 * inv;
    }
}

__global__ __launch_bounds__(128)
void act_route_kernel(const __bf16* __restrict__ xn, const float* __restrict__ w,
                      const float* __restrict__ gumbel, float* __restrict__ route)
{
    __shared__ float red[4][128];
    const int tid = threadIdx.x;
    const long row = blockIdx.x;
    const __bf16* xr = xn + row * VV;
    float s[4] = {0.f, 0.f, 0.f, 0.f};
    for (int i = tid; i < VV; i += 128) {
        const float xv = (float)xr[i];
        s[0] += xv * w[i];
        s[1] += xv * w[VV + i];
        s[2] += xv * w[2 * VV + i];
        s[3] += xv * w[3 * VV + i];
    }
    for (int k = 0; k < 4; ++k) red[k][tid] = s[k];
    __syncthreads();
    for (int o = 64; o > 0; o >>= 1) {
        if (tid < o) for (int k = 0; k < 4; ++k) red[k][tid] += red[k][tid + o];
        __syncthreads();
    }
    if (tid == 0) {
        float l[4], m = -1e30f;
        int am = 0;
        for (int k = 0; k < 4; ++k) {
            l[k] = red[k][0] + gumbel[row * 4 + k];
            if (l[k] > m) { m = l[k]; am = k; }
        }
        float e[4], sum = 0.f;
        for (int k = 0; k < 4; ++k) { e[k] = expf(l[k] - m); sum += e[k]; }
        const float inv = 1.0f / sum;
        for (int k = 0; k < 4; ++k) {
            const float soft = e[k] * inv;
            const float hard = (k == am) ? 1.0f : 0.0f;
            route[row * 4 + k] = (hard - soft) + soft;  // straight-through fwd
        }
    }
}

__global__ void blend_kernel(const float* __restrict__ retr, const float* __restrict__ sw,
                             __bf16* __restrict__ blended, long total, long nstride)
{
    for (long idx = blockIdx.x * (long)blockDim.x + threadIdx.x; idx < total;
         idx += (long)gridDim.x * blockDim.x) {
        const long row = idx / SS;
        const float v = sw[row * 3 + 0] * retr[idx]
                      + sw[row * 3 + 1] * retr[nstride + idx]
                      + sw[row * 3 + 2] * retr[2 * nstride + idx];
        blended[idx] = (__bf16)v;
    }
}

__global__ void residual_kernel(const float* __restrict__ x, const float* __restrict__ d,
                                const float* __restrict__ sc1, const float* __restrict__ sc2,
                                float* __restrict__ out, long total)
{
    const float a = sc1[0] * sc2[0];
    for (long i = blockIdx.x * (long)blockDim.x + threadIdx.x; i < total;
         i += (long)gridDim.x * blockDim.x)
        out[i] = x[i] + a * d[i];
}

__global__ void act_kernel(const float* __restrict__ h, const float* __restrict__ bias,
                           const float* __restrict__ route, __bf16* __restrict__ out, long total)
{
    for (long idx = blockIdx.x * (long)blockDim.x + threadIdx.x; idx < total;
         idx += (long)gridDim.x * blockDim.x) {
        const long row = idx / II;
        const int  i   = (int)(idx - row * II);
        const float hv = h[idx] + bias[i];
        const float* r = route + row * 4;
        const float g  = 0.5f * hv * (1.0f + erff(hv * 0.70710678118654752f));
        const float re = fmaxf(hv, 0.0f);
        const float si = hv / (1.0f + expf(-hv));
        const float th = tanhf(hv);
        out[idx] = (__bf16)(r[0] * g + r[1] * re + r[2] * si + r[3] * th);
    }
}

// ---------------------------------------------------------------------------
// Host launcher
// ---------------------------------------------------------------------------
extern "C" void kernel_launch(void* const* d_in, const int* in_sizes, int n_in,
                              void* d_out, int out_size, void* d_ws, size_t ws_size,
                              hipStream_t stream)
{
    (void)in_sizes; (void)n_in; (void)out_size; (void)ws_size;

    const float* x        = (const float*)d_in[0];
    const float* q_w      = (const float*)d_in[1];
    const float* k_w      = (const float*)d_in[2];
    const float* v_w      = (const float*)d_in[3];
    const float* o_w      = (const float*)d_in[4];
    const float* sg_w     = (const float*)d_in[5];
    const float* dec_l    = (const float*)d_in[6];
    const float* q_os     = (const float*)d_in[7];
    const float* down_w   = (const float*)d_in[8];
    const float* up_w     = (const float*)d_in[9];
    const float* tr_bias  = (const float*)d_in[10];
    const float* sel_w    = (const float*)d_in[11];
    const float* t_os     = (const float*)d_in[12];
    const float* q_scale  = (const float*)d_in[13];
    const float* t_scale  = (const float*)d_in[14];
    const float* gumbel   = (const float*)d_in[15];
    float* out = (float*)d_out;

    // ---- workspace carving ----
    char* base = (char*)d_ws;
    size_t off = 0;
    auto alloc = [&](size_t bytes) -> void* {
        off = (off + 255) & ~(size_t)255;
        void* p = base + off;
        off += bytes;
        return p;
    };
    __bf16* xn1   = (__bf16*)alloc((size_t)BT * VV * 2);
    __bf16* qw_b  = (__bf16*)alloc((size_t)SS * VV * 2);
    __bf16* kw_b  = (__bf16*)alloc((size_t)3 * SS * VV * 2);
    __bf16* vw_b  = (__bf16*)alloc((size_t)3 * SS * VV * 2);
    __bf16* ow_b  = (__bf16*)alloc((size_t)VV * SS * 2);
    __bf16* dw_b  = (__bf16*)alloc((size_t)II * VV * 2);
    __bf16* uw_b  = (__bf16*)alloc((size_t)VV * II * 2);
    __bf16* qb    = (__bf16*)alloc((size_t)BT * SS * 2);
    __bf16* keysb = (__bf16*)alloc((size_t)3 * BT * SS * 2);
    __bf16* valsb = (__bf16*)alloc((size_t)3 * BT * SS * 2);
    float*  sw    = (float*) alloc((size_t)BT * 3 * 4);
    __bf16* scrw  = (__bf16*)alloc((size_t)3 * BB * TT * TT * 2);
    float*  retr  = (float*) alloc((size_t)3 * BT * SS * 4);
    __bf16* blnd  = (__bf16*)alloc((size_t)BT * SS * 2);
    float*  oproj = (float*) alloc((size_t)BT * VV * 4);
    float*  x1    = (float*) alloc((size_t)BT * VV * 4);
    __bf16* xn2   = (__bf16*)alloc((size_t)BT * VV * 2);
    float*  hbuf  = (float*) alloc((size_t)BT * II * 4);
    float*  route = (float*) alloc((size_t)BT * 4 * 4);
    __bf16* actb  = (__bf16*)alloc((size_t)BT * II * 2);
    float*  upbuf = (float*) alloc((size_t)BT * VV * 4);

    const dim3 blk256(256);
    const int GS = 4096;  // grid-stride blocks

    // 1) rms_norm(x) -> bf16
    rmsnorm_bf16_kernel<<<BT, 256, 0, stream>>>(x, xn1);

    // 2) weight converts
    f32_to_bf16_kernel<<<GS, 256, 0, stream>>>(q_w,    qw_b, (long)SS * VV);
    f32_to_bf16_kernel<<<GS, 256, 0, stream>>>(k_w,    kw_b, (long)3 * SS * VV);
    f32_to_bf16_kernel<<<GS, 256, 0, stream>>>(v_w,    vw_b, (long)3 * SS * VV);
    f32_to_bf16_kernel<<<GS, 256, 0, stream>>>(o_w,    ow_b, (long)VV * SS);
    f32_to_bf16_kernel<<<GS, 256, 0, stream>>>(down_w, dw_b, (long)II * VV);
    f32_to_bf16_kernel<<<GS, 256, 0, stream>>>(up_w,   uw_b, (long)VV * II);

    // 3) q = xn @ q_w^T  [BT,S], bf16 out
    gemm_wmma<1, false><<<dim3(SS / 128, BT / 128, 1), blk256, 0, stream>>>(
        xn1, qw_b, qb, BT, SS, VV, 0, 0, 0, 0, 0, 0, 1, nullptr);

    // 4) keys / vals (3 heads)
    gemm_wmma<1, false><<<dim3(SS / 128, BT / 128, 3), blk256, 0, stream>>>(
        xn1, kw_b, keysb, BT, SS, VV,
        0, 0, (long)SS * VV, 0, (long)BT * SS, 0, 1, nullptr);
    gemm_wmma<1, false><<<dim3(SS / 128, BT / 128, 3), blk256, 0, stream>>>(
        xn1, vw_b, valsb, BT, SS, VV,
        0, 0, (long)SS * VV, 0, (long)BT * SS, 0, 1, nullptr);

    // 5) softmax gate
    scale_gate_kernel<<<BT, 128, 0, stream>>>(xn1, sg_w, sw);

    // 6) scores = q @ keys^T with fused decay mask -> bf16 [3,B,T,T]
    gemm_wmma<2, false><<<dim3(TT / 128, TT / 128, 12), blk256, 0, stream>>>(
        qb, keysb, scrw, TT, TT, SS,
        0, (long)TT * SS,                    // A (q):     per-b stride
        (long)BT * SS, (long)TT * SS,        // B (keys):  per-n, per-b
        (long)BB * TT * TT, (long)TT * TT,   // C:         per-n, per-b
        BB, dec_l);

    // 7) retrieved = scores_w @ vals  (B is [K=T, N=S] row-major)
    gemm_wmma<0, true><<<dim3(SS / 128, TT / 128, 12), blk256, 0, stream>>>(
        scrw, valsb, retr, TT, SS, TT,
        (long)BB * TT * TT, (long)TT * TT,
        (long)BT * SS, (long)TT * SS,
        (long)BT * SS, (long)TT * SS,
        BB, nullptr);

    // 8) blend heads
    blend_kernel<<<GS, 256, 0, stream>>>(retr, sw, blnd, (long)BT * SS, (long)BT * SS);

    // 9) o_proj: blended @ o_w^T  [BT,V] fp32
    gemm_wmma<0, false><<<dim3(VV / 128, BT / 128, 1), blk256, 0, stream>>>(
        blnd, ow_b, oproj, BT, VV, SS, 0, 0, 0, 0, 0, 0, 1, nullptr);

    // 10) x1 = x + q_scale*q_out_scale*oproj
    residual_kernel<<<GS, 256, 0, stream>>>(x, oproj, q_scale, q_os, x1, (long)BT * VV);

    // 11) rms_norm(x1) -> bf16
    rmsnorm_bf16_kernel<<<BT, 256, 0, stream>>>(x1, xn2);

    // 12) gumbel routing
    act_route_kernel<<<BT, 128, 0, stream>>>(xn2, sel_w, gumbel, route);

    // 13) h = xn2 @ down_w^T  [BT,I] fp32
    gemm_wmma<0, false><<<dim3(II / 128, BT / 128, 1), blk256, 0, stream>>>(
        xn2, dw_b, hbuf, BT, II, VV, 0, 0, 0, 0, 0, 0, 1, nullptr);

    // 14) routed activation -> bf16
    act_kernel<<<GS, 256, 0, stream>>>(hbuf, tr_bias, route, actb, (long)BT * II);

    // 15) up: act @ up_w^T  [BT,V] fp32
    gemm_wmma<0, false><<<dim3(VV / 128, BT / 128, 1), blk256, 0, stream>>>(
        actb, uw_b, upbuf, BT, VV, II, 0, 0, 0, 0, 0, 0, 1, nullptr);

    // 16) out = x1 + t_scale*t_out_scale*up
    residual_kernel<<<GS, 256, 0, stream>>>(x1, upbuf, t_scale, t_os, out, (long)BT * VV);
}
